// BottleNeck_MHSA_63299228008588
// MI455X (gfx1250) — matrix operand order repaired and checked by reading, not compiled, not run on hardware
//
#include <hip/hip_runtime.h>

// MI455X / gfx1250, wave32. bf16 WMMA (16x16x32) with f32 accumulate.

typedef __attribute__((ext_vector_type(16))) __bf16 bf16x16;
typedef __attribute__((ext_vector_type(8)))  __bf16 bf16x8;
typedef __attribute__((ext_vector_type(8)))  float  f32x8;

#define DEV __device__ __forceinline__

constexpr int Bc  = 16;
constexpr int Cc  = 512;
constexpr int Nc  = 1024;   // H*W
constexpr int NHc = 8;
constexpr int DKc = 64;

DEV __bf16 f2bf(float f) {
  union { float f; unsigned u; } v; v.f = f;
  unsigned r = (v.u + 0x7FFFu + ((v.u >> 16) & 1u)) >> 16;   // RNE
  unsigned short s = (unsigned short)r;
  __bf16 b; __builtin_memcpy(&b, &s, 2); return b;
}

#define WMMA_BF16(a, b, c) \
  __builtin_amdgcn_wmma_f32_16x16x32_bf16(false, (a), false, (b), (short)0, (c), false, false)

// A fragment: 16x32 (MxK), row-major source. lanes 0-15 row M=lane: K {0..7,16..23},
// lanes 16-31 same rows: K {8..15,24..31}.
DEV bf16x16 load_fragA(const __bf16* __restrict__ src, int lda, int row0, int k0) {
  int lane = threadIdx.x & 31;
  const __bf16* p = src + (size_t)(row0 + (lane & 15)) * lda + k0 + ((lane >> 4) << 3);
  bf16x8 lo = *(const bf16x8*)p;
  bf16x8 hi = *(const bf16x8*)(p + 16);
  bf16x16 r;
#pragma unroll
  for (int i = 0; i < 8; ++i) { r[i] = lo[i]; r[i + 8] = hi[i]; }
  return r;
}

// B fragment: 32x16 (KxN) where column n == row (col0+n) of row-major src (B = srcT).
// lanes 0-15: N=lane, K {0..7,16..23}; lanes 16-31: N=lane-16, K {8..15,24..31}.
DEV bf16x16 load_fragB(const __bf16* __restrict__ src, int ldb, int col0, int k0) {
  int lane = threadIdx.x & 31;
  const __bf16* p = src + (size_t)(col0 + (lane & 15)) * ldb + k0 + ((lane >> 4) << 3);
  bf16x8 lo = *(const bf16x8*)p;
  bf16x8 hi = *(const bf16x8*)(p + 16);
  bf16x16 r;
#pragma unroll
  for (int i = 0; i < 8; ++i) { r[i] = lo[i]; r[i + 8] = hi[i]; }
  return r;
}

DEV float rowmax16(float v) {
  v = fmaxf(v, __shfl_xor(v, 1, 32));
  v = fmaxf(v, __shfl_xor(v, 2, 32));
  v = fmaxf(v, __shfl_xor(v, 4, 32));
  v = fmaxf(v, __shfl_xor(v, 8, 32));
  return v;
}
DEV float rowsum16(float v) {
  v += __shfl_xor(v, 1, 32);
  v += __shfl_xor(v, 2, 32);
  v += __shfl_xor(v, 4, 32);
  v += __shfl_xor(v, 8, 32);
  return v;
}

// ---------------- pack kernels ----------------

// Xt[b][p][c] = x[b][c][p] via 32x32 LDS tile transpose (coalesced both sides).
__global__ void __launch_bounds__(256) k_pack_xt(const float* __restrict__ x,
                                                 __bf16* __restrict__ xt) {
  __shared__ float ld[32][33];
  int tb = blockIdx.x;              // Bc * 512 tiles
  int b  = tb >> 9;
  int t  = tb & 511;
  int c0 = (t >> 5) << 5;           // 16 channel tiles
  int p0 = (t & 31) << 5;           // 32 pixel tiles
  int row = threadIdx.x >> 3;       // 0..31
  int col = (threadIdx.x & 7) << 2; // 0,4,..,28
  const float* src = x + ((size_t)b * Cc + c0 + row) * Nc + p0 + col;
  float4 v = *(const float4*)src;
  ld[row][col] = v.x; ld[row][col + 1] = v.y; ld[row][col + 2] = v.z; ld[row][col + 3] = v.w;
  __syncthreads();
  __bf16* dst = xt + ((size_t)b * Nc + p0 + row) * Cc + c0 + col;
#pragma unroll
  for (int i = 0; i < 4; ++i) dst[i] = f2bf(ld[col + i][row]);
}

__global__ void k_pack_mat(const float* __restrict__ src, __bf16* __restrict__ dst, int n) {
  int e = blockIdx.x * blockDim.x + threadIdx.x;
  if (e < n) dst[e] = f2bf(src[e]);
}

// relpad[t][r][d]: t=0 rel_h, t=1 rel_w; rows 0..62 copied, row 63 zero.
__global__ void k_pack_rel(const float* __restrict__ relh, const float* __restrict__ relw,
                           __bf16* __restrict__ dst) {
  int e = blockIdx.x * blockDim.x + threadIdx.x;
  if (e >= 2 * 64 * 64) return;
  int t = e >> 12;
  int rr = (e >> 6) & 63;
  int d = e & 63;
  float v = (rr < 63) ? (t ? relw[rr * 64 + d] : relh[rr * 64 + d]) : 0.f;
  dst[e] = f2bf(v);
}

// ---------------- QKV projection: Yt(NxC) = Xt(NxC) @ W^T per batch ----------------
// wave: 32 pixels x 64 out-channels (2x4 register blocking, 8 accumulators), K = 512.
__global__ void __launch_bounds__(256) k_qkv(
    const __bf16* __restrict__ xt, const __bf16* __restrict__ wq,
    const __bf16* __restrict__ wk, const __bf16* __restrict__ wv,
    __bf16* __restrict__ qb, __bf16* __restrict__ kb, __bf16* __restrict__ vt) {
  int wid  = (blockIdx.x * blockDim.x + threadIdx.x) >> 5;   // 12288 waves
  int lane = threadIdx.x & 31, hf = lane >> 4, ln = lane & 15;
  int b    = wid / 768;
  int rem  = wid % 768;
  int proj = rem >> 8;
  int t    = rem & 255;
  int p0 = (t >> 3) << 5;      // 32-pixel tile
  int o0 = (t & 7) << 6;       // 64 out-channels
  const __bf16* X  = xt + (size_t)b * Nc * Cc;
  const __bf16* Wm = (proj == 0) ? wq : (proj == 1) ? wk : wv;

  f32x8 acc0[4] = {}, acc1[4] = {};
#pragma unroll
  for (int ks = 0; ks < 16; ++ks) {
    int k0 = ks * 32;
    bf16x16 a0 = load_fragA(X, Cc, p0, k0);
    bf16x16 a1 = load_fragA(X, Cc, p0 + 16, k0);
#pragma unroll
    for (int j = 0; j < 4; ++j) {
      bf16x16 bf = load_fragB(Wm, Cc, o0 + j * 16, k0);
      acc0[j] = WMMA_BF16(a0, bf, acc0[j]);
      acc1[j] = WMMA_BF16(a1, bf, acc1[j]);
    }
  }
  // scatter to head layouts: o = dk*NH + nh  ->  nh = o&7, dk = o>>3
#pragma unroll
  for (int h = 0; h < 2; ++h) {
    int pb = p0 + h * 16;
#pragma unroll
    for (int j = 0; j < 4; ++j) {
      int o = o0 + j * 16 + ln, nh = o & 7, dk = o >> 3;
#pragma unroll
      for (int vi = 0; vi < 8; ++vi) {
        int p = pb + vi + hf * 8;
        __bf16 val = f2bf(h ? acc1[j][vi] : acc0[j][vi]);
        if (proj == 0)      qb[(((size_t)b * NHc + nh) * Nc + p) * DKc + dk] = val;
        else if (proj == 1) kb[(((size_t)b * NHc + nh) * Nc + p) * DKc + dk] = val;
        else                vt[(((size_t)b * NHc + nh) * DKc + dk) * Nc + p] = val;
      }
    }
  }
}

// ---------------- rel-pos logits: LH/LW (B,NH,N,64) f32 = Q @ relpad^T ----------------
__global__ void __launch_bounds__(256) k_lhlw(
    const __bf16* __restrict__ qb, const __bf16* __restrict__ relpad,
    float* __restrict__ lh, float* __restrict__ lw) {
  int wid  = (blockIdx.x * blockDim.x + threadIdx.x) >> 5;   // 16384 waves
  int lane = threadIdx.x & 31, hf = lane >> 4, ln = lane & 15;
  int t   = wid >> 13;               // 0 -> LH (rel_h), 1 -> LW (rel_w)
  int r2  = wid & 8191;
  int b   = r2 >> 9;
  int rem = r2 & 511;
  int nh  = rem >> 6;
  int p0  = (rem & 63) << 4;
  const __bf16* Q = qb + ((size_t)b * NHc + nh) * Nc * DKc;
  const __bf16* R = relpad + t * 4096;
  float* dst = (t ? lw : lh) + ((size_t)b * NHc + nh) * Nc * 64;

  f32x8 acc[4] = {};
#pragma unroll
  for (int ks = 0; ks < 2; ++ks) {
    bf16x16 a = load_fragA(Q, DKc, p0, ks * 32);
#pragma unroll
    for (int j = 0; j < 4; ++j) {
      bf16x16 bf = load_fragB(R, 64, j * 16, ks * 32);
      acc[j] = WMMA_BF16(a, bf, acc[j]);
    }
  }
#pragma unroll
  for (int j = 0; j < 4; ++j)
#pragma unroll
    for (int vi = 0; vi < 8; ++vi)
      dst[(size_t)(p0 + vi + hf * 8) * 64 + j * 16 + ln] = acc[j][vi];
}

// ---------------- fused streaming attention (flash-style, online softmax) ----------------
// wave: 16 queries x DK=64 output, streaming 1024 keys in 32-key chunks.
// Bias terms are register-resident: LW term is chunk-invariant; LH term is one value
// per (row, chunk) fetched from lane-distributed registers via __shfl (bpermute).
__global__ void __launch_bounds__(256) k_attn(
    const __bf16* __restrict__ qb, const __bf16* __restrict__ kb,
    const __bf16* __restrict__ vt, const float* __restrict__ lh,
    const float* __restrict__ lw, __bf16* __restrict__ op) {
  __shared__ __bf16 sP[8 * 16 * 32];          // per-wave 16x32 P tile (8KB)
  int wid   = (blockIdx.x * blockDim.x + threadIdx.x) >> 5;
  int wslot = (threadIdx.x >> 5) & 7;
  int lane  = threadIdx.x & 31, hf = lane >> 4, ln = lane & 15;
  int b   = wid >> 9;
  int rem = wid & 511;
  int nh  = rem >> 6;
  int p0  = (rem & 63) << 4;
  size_t hb = (size_t)b * NHc + nh;
  const __bf16* Q  = qb + hb * Nc * DKc;
  const __bf16* K  = kb + hb * Nc * DKc;
  const __bf16* V  = vt + hb * DKc * Nc;      // (DK x N)
  const float*  LH = lh + hb * Nc * 64;
  const float*  LW = lw + hb * Nc * 64;

  bf16x16 a0 = load_fragA(Q, DKc, p0, 0);
  bf16x16 a1 = load_fragA(Q, DKc, p0, 32);

  int prow[8];
  float lh_preA[8], lh_preB[8], lwv0[8], lwv1[8];
#pragma unroll
  for (int vi = 0; vi < 8; ++vi) {
    int p = p0 + vi + hf * 8;
    prow[vi] = p;
    int xq = p >> 5, yq = p & 31;
    // lane l of each half holds LH for key-row i = l (A) and i = 16+l (B) of its rows
    lh_preA[vi] = LH[(size_t)p * 64 + (ln      - xq + 31)];
    lh_preB[vi] = LH[(size_t)p * 64 + (ln + 16 - xq + 31)];
    // LW term: key%32 = t*16 + ln is the same every chunk -> fully loop-invariant
    lwv0[vi]    = LW[(size_t)p * 64 + (ln      - yq + 31)];
    lwv1[vi]    = LW[(size_t)p * 64 + (ln + 16 - yq + 31)];
  }

  float M[8], L[8];
  f32x8 Oacc[4] = {};
#pragma unroll
  for (int vi = 0; vi < 8; ++vi) { M[vi] = -1e30f; L[vi] = 0.f; }

  for (int kc = 0; kc < Nc; kc += 32) {
    if (kc + 32 < Nc) __builtin_prefetch(K + (size_t)(kc + 32) * DKc, 0, 3);

    // LH value for this chunk's key-row i, per query row (bpermute from lane i of my half)
    int i = kc >> 5;                       // uniform
    float lhc[8];
#pragma unroll
    for (int vi = 0; vi < 8; ++vi) {
      float srcv = (i < 16) ? lh_preA[vi] : lh_preB[vi];
      lhc[vi] = __shfl(srcv, (hf << 4) | (i & 15), 32);
    }

    // energy tiles
    float s[2][8];
#pragma unroll
    for (int t = 0; t < 2; ++t) {
      int c0 = kc + t * 16;
      bf16x16 b0 = load_fragB(K, DKc, c0, 0);
      bf16x16 b1 = load_fragB(K, DKc, c0, 32);
      f32x8 e = {};
      e = WMMA_BF16(a0, b0, e);
      e = WMMA_BF16(a1, b1, e);
#pragma unroll
      for (int vi = 0; vi < 8; ++vi)
        s[t][vi] = (e[vi] + lhc[vi] + (t ? lwv1[vi] : lwv0[vi])) * 0.125f;  // 1/sqrt(DK)
    }

    // hoist V fragments so global loads overlap the softmax VALU work
    bf16x16 vb0 = load_fragB(V, Nc, 0,  kc);
    bf16x16 vb1 = load_fragB(V, Nc, 16, kc);
    bf16x16 vb2 = load_fragB(V, Nc, 32, kc);
    bf16x16 vb3 = load_fragB(V, Nc, 48, kc);

    // online softmax update (rows live across 16-lane halves)
#pragma unroll
    for (int vi = 0; vi < 8; ++vi) {
      float mn = rowmax16(fmaxf(s[0][vi], s[1][vi]));
      mn = fmaxf(M[vi], mn);
      float alpha = __expf(M[vi] - mn);
      M[vi] = mn;
      float e0 = __expf(s[0][vi] - mn);
      float e1 = __expf(s[1][vi] - mn);
      L[vi] = L[vi] * alpha + rowsum16(e0 + e1);
      int row = vi + hf * 8;
      sP[wslot * 512 + row * 32 + ln]      = f2bf(e0);
      sP[wslot * 512 + row * 32 + 16 + ln] = f2bf(e1);
#pragma unroll
      for (int j = 0; j < 4; ++j) Oacc[j][vi] *= alpha;
    }
    __syncthreads();                        // C/D -> A layout transpose through LDS
    const __bf16* pp = &sP[wslot * 512 + ln * 32 + hf * 8];
    bf16x8 lo = *(const bf16x8*)pp;
    bf16x8 hi = *(const bf16x8*)(pp + 16);
    bf16x16 pa;
#pragma unroll
    for (int i2 = 0; i2 < 8; ++i2) { pa[i2] = lo[i2]; pa[i2 + 8] = hi[i2]; }
    Oacc[0] = WMMA_BF16(pa, vb0, Oacc[0]);
    Oacc[1] = WMMA_BF16(pa, vb1, Oacc[1]);
    Oacc[2] = WMMA_BF16(pa, vb2, Oacc[2]);
    Oacc[3] = WMMA_BF16(pa, vb3, Oacc[3]);
    __syncthreads();
  }

#pragma unroll
  for (int vi = 0; vi < 8; ++vi) L[vi] = 1.f / L[vi];
  // store attention output pixel-major: Op[b][p][c'] with c' = nh*64 + dk
#pragma unroll
  for (int j = 0; j < 4; ++j) {
    int dk = j * 16 + ln;
#pragma unroll
    for (int vi = 0; vi < 8; ++vi)
      op[((size_t)b * Nc + prow[vi]) * Cc + nh * DKc + dk] = f2bf(Oacc[j][vi] * L[vi]);
  }
}

// ---------------- output projection: out[b][o][p] = (Op @ Wo^T)[p][o] + b_o[o] ----------------
__global__ void __launch_bounds__(256) k_oproj(
    const __bf16* __restrict__ op, const __bf16* __restrict__ wo,
    const float* __restrict__ bo, float* __restrict__ out) {
  int wid  = (blockIdx.x * blockDim.x + threadIdx.x) >> 5;   // 4096 waves
  int lane = threadIdx.x & 31, hf = lane >> 4, ln = lane & 15;
  int b   = wid >> 8;
  int rem = wid & 255;
  int p0  = (rem >> 3) << 5;
  int o0  = (rem & 7) << 6;
  const __bf16* X = op + (size_t)b * Nc * Cc;

  f32x8 acc0[4] = {}, acc1[4] = {};
#pragma unroll
  for (int ks = 0; ks < 16; ++ks) {
    int k0 = ks * 32;
    bf16x16 a0 = load_fragA(X, Cc, p0, k0);
    bf16x16 a1 = load_fragA(X, Cc, p0 + 16, k0);
#pragma unroll
    for (int j = 0; j < 4; ++j) {
      bf16x16 bf = load_fragB(wo, Cc, o0 + j * 16, k0);
      acc0[j] = WMMA_BF16(a0, bf, acc0[j]);
      acc1[j] = WMMA_BF16(a1, bf, acc1[j]);
    }
  }
#pragma unroll
  for (int h = 0; h < 2; ++h) {
    int pb = p0 + h * 16;
#pragma unroll
    for (int j = 0; j < 4; ++j) {
      int o = o0 + j * 16 + ln;
      float bias = bo[o];
#pragma unroll
      for (int vi = 0; vi < 8; ++vi) {
        int p = pb + vi + hf * 8;
        out[((size_t)b * Cc + o) * Nc + p] = (h ? acc1[j][vi] : acc0[j][vi]) + bias;
      }
    }
  }
}

// ---------------- host ----------------

extern "C" void kernel_launch(void* const* d_in, const int* in_sizes, int n_in,
                              void* d_out, int out_size, void* d_ws, size_t ws_size,
                              hipStream_t stream) {
  (void)in_sizes; (void)n_in; (void)out_size; (void)ws_size;
  const float* x    = (const float*)d_in[0];
  const float* w_q  = (const float*)d_in[1];
  const float* w_k  = (const float*)d_in[2];
  const float* w_v  = (const float*)d_in[3];
  const float* w_o  = (const float*)d_in[4];
  const float* b_o  = (const float*)d_in[5];
  const float* relh = (const float*)d_in[6];
  const float* relw = (const float*)d_in[7];
  float* out = (float*)d_out;

  char* ws = (char*)d_ws;
  size_t off = 0;
  auto alloc = [&](size_t bytes) { char* p = ws + off; off += (bytes + 255) & ~(size_t)255; return p; };
  __bf16* XT  = (__bf16*)alloc((size_t)Bc * Nc * Cc * 2);
  __bf16* WQ  = (__bf16*)alloc((size_t)Cc * Cc * 2);
  __bf16* WK  = (__bf16*)alloc((size_t)Cc * Cc * 2);
  __bf16* WV  = (__bf16*)alloc((size_t)Cc * Cc * 2);
  __bf16* WO  = (__bf16*)alloc((size_t)Cc * Cc * 2);
  __bf16* REL = (__bf16*)alloc((size_t)2 * 64 * 64 * 2);
  __bf16* QB  = (__bf16*)alloc((size_t)Bc * NHc * Nc * DKc * 2);
  __bf16* KB  = (__bf16*)alloc((size_t)Bc * NHc * Nc * DKc * 2);
  __bf16* VT  = (__bf16*)alloc((size_t)Bc * NHc * DKc * Nc * 2);
  float*  LHt = (float*) alloc((size_t)Bc * NHc * Nc * 64 * 4);
  float*  LWt = (float*) alloc((size_t)Bc * NHc * Nc * 64 * 4);
  __bf16* OP  = (__bf16*)alloc((size_t)Bc * Nc * Cc * 2);

  const int TB = 256;
  // pack
  k_pack_xt<<<Bc * 512, TB, 0, stream>>>(x, XT);
  k_pack_mat<<<(Cc * Cc + TB - 1) / TB, TB, 0, stream>>>(w_q, WQ, Cc * Cc);
  k_pack_mat<<<(Cc * Cc + TB - 1) / TB, TB, 0, stream>>>(w_k, WK, Cc * Cc);
  k_pack_mat<<<(Cc * Cc + TB - 1) / TB, TB, 0, stream>>>(w_v, WV, Cc * Cc);
  k_pack_mat<<<(Cc * Cc + TB - 1) / TB, TB, 0, stream>>>(w_o, WO, Cc * Cc);
  k_pack_rel<<<(2 * 64 * 64 + TB - 1) / TB, TB, 0, stream>>>(relh, relw, REL);
  // QKV projections: 16 batches * 3 proj * (32 pixel-tiles * 8 col-tiles) waves
  k_qkv<<<(Bc * 3 * 256 * 32) / TB, TB, 0, stream>>>(XT, WQ, WK, WV, QB, KB, VT);
  // rel-pos logit tables
  k_lhlw<<<(2 * Bc * NHc * 64 * 32) / TB, TB, 0, stream>>>(QB, REL, LHt, LWt);
  // fused attention
  k_attn<<<(Bc * NHc * 64 * 32) / TB, TB, 0, stream>>>(QB, KB, VT, LHt, LWt, OP);
  // output projection
  k_oproj<<<(Bc * 256 * 32) / TB, TB, 0, stream>>>(OP, WO, b_o, out);
}
